// ANDP_61847529063010
// MI455X (gfx1250) — matrix operand order repaired and checked by reading, not compiled, not run on hardware
//
#include <hip/hip_runtime.h>
#include <hip/hip_bf16.h>

typedef __attribute__((ext_vector_type(16))) _Float16 v16h;
typedef __attribute__((ext_vector_type(8)))  float    v8f;
typedef __attribute__((ext_vector_type(4)))  unsigned v4u;
typedef __attribute__((ext_vector_type(8)))  int      v8i;
typedef __attribute__((ext_vector_type(4)))  int      v4i;

#define D_DIM 16
#define N_SYS 32
#define H_DIM 128

// Workspace fragment offsets (in _Float16 elements). Each fragment = 32 lanes * 16 halves = 512.
#define OFF_B1 0        // GEMM1 B (W1^T, K padded 16->32): 8 frags
#define OFF_B2 4096     // GEMM2 B (W2^T): 2 n-tiles x 4 k-tiles = 8 frags
#define OFF_B3 8192     // GEMM3 B (A_flat[(n,d),k]): 16 k-tiles
#define WS_HALFS 16384  // 32 KB total

#ifndef __has_builtin
#define __has_builtin(x) 0
#endif
#if __has_builtin(__builtin_amdgcn_tensor_load_to_lds) && \
    __has_builtin(__builtin_amdgcn_s_wait_tensorcnt)
#define USE_TDM 1
#else
#define USE_TDM 0
#endif

// ---------------------------------------------------------------------------
// Prep: pack weights into wave32 WMMA B-fragment layout (ISA 7.12.2):
//   fragment element (lane, h):  K = (h&7) + 8*(lane>>4) + 16*(h>>3),  N = lane&15
// ---------------------------------------------------------------------------
__global__ __launch_bounds__(256) void ANDP_prep_frags(
    const float* __restrict__ W1, const float* __restrict__ W2,
    const float* __restrict__ Bm, const float* __restrict__ Cm,
    _Float16* __restrict__ ws)
{
  int e = blockIdx.x * blockDim.x + threadIdx.x;
  if (e >= WS_HALFS) return;
  int r      = e & 511;
  int lane   = r >> 4;
  int h      = r & 15;
  int laneHi = lane >> 4;
  int laneN  = lane & 15;
  float v;
  if (e < OFF_B2) {                       // B1[K][N] = W1[N][K], zero-pad K>=16
    int frag = e >> 9;
    int K = (h & 7) + 8 * laneHi + 16 * (h >> 3);
    int n = frag * 16 + laneN;
    v = (K < D_DIM) ? W1[n * D_DIM + K] : 0.0f;
  } else if (e < OFF_B3) {                // B2[K][N] = W2[N][K]
    int frag = (e - OFF_B2) >> 9;
    int nt = frag >> 2, kt = frag & 3;
    int K = kt * 32 + (h & 7) + 8 * laneHi + 16 * (h >> 3);
    int n = nt * 16 + laneN;
    v = W2[n * H_DIM + K];
  } else {                                // B3[(n,d)][k] = Bm[n,k,d] + Cm[n,k,d]
    int frag = (e - OFF_B3) >> 9;
    int Kf = frag * 32 + (h & 7) + 8 * laneHi + 16 * (h >> 3);
    int n = Kf >> 4, d = Kf & 15;
    int k = laneN;
    int idx = (n * D_DIM + k) * D_DIM + d;
    v = Bm[idx] + Cm[idx];
  }
  ws[e] = (_Float16)v;
}

// ---------------------------------------------------------------------------
// Main fused kernel: each wave32 owns one 16-row tile per loop iteration.
// Weight fragments staged in LDS once per block via the Tensor Data Mover.
// ---------------------------------------------------------------------------
__global__ __launch_bounds__(256, 6) void ANDP_main(
    const float* __restrict__ x,  const float* __restrict__ b1,
    const float* __restrict__ b2, const float* __restrict__ xt,
    const _Float16* __restrict__ ws, float* __restrict__ out, int numTiles)
{
  __shared__ __align__(32) _Float16 wtab[WS_HALFS];     // 32 KB pre-swizzled weight frags
  __shared__ __align__(16) _Float16 hLDS[8][16 * H_DIM];  // 32 KB h activations per wave
  __shared__ float    lgLDS[8][16 * N_SYS];               // 16 KB logits per wave
  __shared__ _Float16 wLDS[8][16 * N_SYS];                //  8 KB softmax weights per wave

  const int wave   = threadIdx.x >> 5;   // wave32
  const int lane   = threadIdx.x & 31;
  const int laneM  = lane & 15;
  const int laneHi = lane >> 4;
  const int gwave  = blockIdx.x * 8 + wave;
  const int nWaves = gridDim.x * 8;

  // ---- stage the 32 KB weight-fragment table into LDS ----
#if USE_TDM
  if (threadIdx.x < 32) {   // wave 0 issues one TDM DMA for the whole table
    unsigned ldsAddr = (unsigned)(unsigned long long)(&wtab[0]);
    unsigned long long ga = (unsigned long long)ws;
    v4u g0 = {0u, 0u, 0u, 0u};
    g0[0] = 1u;                                       // count=1, user descriptor
    g0[1] = ldsAddr;                                  // lds_addr [63:32]
    g0[2] = (unsigned)(ga & 0xffffffffu);             // global_addr [95:64]
    g0[3] = (unsigned)((ga >> 32) & 0x01ffffffu)      // global_addr [120:96]
          | (2u << 30);                               // type=2 ("image")
    v8i g1 = {0, 0, 0, 0, 0, 0, 0, 0};
    g1[0] = (1 << 16);                  // data_size=1 (2 bytes), mask=0, no pad/iter
    g1[1] = (int)(16384u << 16);        // tensor_dim0[15:0] = 16384 in bits [63:48]
    g1[2] = (1 << 16);                  // tensor_dim1 = 1 (bits [111:80] low half)
    g1[3] = (int)(16384u << 16);        // tile_dim0 = 16384 (bits [127:112])
    g1[4] = 1;                          // tile_dim1 = 1, tile_dim2 = 0
    g1[5] = 16384;                      // tensor_dim0_stride (bits [191:160])
    g1[6] = 0;                          // stride hi, tensor_dim1_stride lo
    g1[7] = 0;
    v4i gz = {0, 0, 0, 0};
#if __clang_major__ >= 23
    v8i gz8 = {0, 0, 0, 0, 0, 0, 0, 0};
    __builtin_amdgcn_tensor_load_to_lds(g0, g1, gz, gz, gz8, 0);
#else
    __builtin_amdgcn_tensor_load_to_lds(g0, g1, gz, gz, 0);
#endif
    __builtin_amdgcn_s_wait_tensorcnt(0);
  }
#else
  for (int i = threadIdx.x; i < WS_HALFS / 8; i += 256)   // uint4 = 8 halves
    ((uint4*)wtab)[i] = ((const uint4*)ws)[i];
#endif
  __syncthreads();

  for (int tile = gwave; tile < numTiles; tile += nWaves) {
    const int b0 = tile * 16;
    if (tile + nWaves < numTiles)        // speculative prefetch of next tile's x rows
      __builtin_prefetch(x + (size_t)(b0 + nWaves * 16 + laneM) * D_DIM, 0, 1);

    // ---- load x tile straight into A-fragment layout: lane -> row M, cols laneHi*8..+8
    const float* xrow = x + (size_t)(b0 + laneM) * D_DIM + laneHi * 8;
    float4 xa = *(const float4*)xrow;
    float4 xb = *(const float4*)(xrow + 4);

    v16h a1;
    a1[0]=(_Float16)xa.x; a1[1]=(_Float16)xa.y; a1[2]=(_Float16)xa.z; a1[3]=(_Float16)xa.w;
    a1[4]=(_Float16)xb.x; a1[5]=(_Float16)xb.y; a1[6]=(_Float16)xb.z; a1[7]=(_Float16)xb.w;
#pragma unroll
    for (int i = 8; i < 16; ++i) a1[i] = (_Float16)0.0f;   // K 16..31 zero pad

    // diff = x_target - x, same 8-col window this lane will need for U fragments
    _Float16 dh[8];
    {
      const float* xtp = xt + laneHi * 8;
      dh[0]=(_Float16)(xtp[0]-xa.x); dh[1]=(_Float16)(xtp[1]-xa.y);
      dh[2]=(_Float16)(xtp[2]-xa.z); dh[3]=(_Float16)(xtp[3]-xa.w);
      dh[4]=(_Float16)(xtp[4]-xb.x); dh[5]=(_Float16)(xtp[5]-xb.y);
      dh[6]=(_Float16)(xtp[6]-xb.z); dh[7]=(_Float16)(xtp[7]-xb.w);
    }

    // ---- GEMM1: h = relu(x @ W1^T + b1), 8 WMMAs, write f16 h to LDS ----
#pragma unroll
    for (int nt = 0; nt < 8; ++nt) {
      v16h bf = *(const v16h*)(wtab + OFF_B1 + nt * 512 + lane * 16);
      v8f acc = {};
      acc = __builtin_amdgcn_wmma_f32_16x16x32_f16(false, a1, false, bf,
                                                   (short)0, acc, false, false);
      float bias = b1[nt * 16 + laneM];        // C-layout: N = lane&15 fixed per lane
#pragma unroll
      for (int r2 = 0; r2 < 8; ++r2) {
        float v = acc[r2] + bias;
        v = v > 0.f ? v : 0.f;
        hLDS[wave][(r2 + laneHi * 8) * H_DIM + nt * 16 + laneM] = (_Float16)v;
      }
    }
    __builtin_amdgcn_wave_barrier();

    // ---- GEMM2: logits = h @ W2^T + b2, 8 WMMAs (A from LDS transpose) ----
#pragma unroll
    for (int nt2 = 0; nt2 < 2; ++nt2) {
      v8f acc = {};
#pragma unroll
      for (int kt = 0; kt < 4; ++kt) {
        const _Float16* ab = &hLDS[wave][laneM * H_DIM + kt * 32 + laneHi * 8];
        union { v16h v; uint4 q[2]; } a2;
        a2.q[0] = *(const uint4*)(ab);         // halves 0..7  : K = kt*32 + laneHi*8 + h
        a2.q[1] = *(const uint4*)(ab + 16);    // halves 8..15 : K += 16
        v16h bf = *(const v16h*)(wtab + OFF_B2 + (nt2 * 4 + kt) * 512 + lane * 16);
        acc = __builtin_amdgcn_wmma_f32_16x16x32_f16(false, a2.v, false, bf,
                                                     (short)0, acc, false, false);
      }
      float bias = b2[nt2 * 16 + laneM];
#pragma unroll
      for (int r2 = 0; r2 < 8; ++r2)
        lgLDS[wave][(r2 + laneHi * 8) * N_SYS + nt2 * 16 + laneM] = acc[r2] + bias;
    }
    __builtin_amdgcn_wave_barrier();

    // ---- row softmax: lane L<16 owns row L (32 logits), write f16 weights ----
    if (lane < 16) {
      const float* Lr = &lgLDS[wave][lane * N_SYS];
      float mx = Lr[0];
#pragma unroll
      for (int n = 1; n < N_SYS; ++n) mx = fmaxf(mx, Lr[n]);
      float s = 0.f;
#pragma unroll
      for (int n = 0; n < N_SYS; ++n) s += __expf(Lr[n] - mx);
      float inv = 1.f / s;
#pragma unroll
      for (int n = 0; n < N_SYS; ++n)
        wLDS[wave][lane * N_SYS + n] = (_Float16)(__expf(Lr[n] - mx) * inv);
    }
    __builtin_amdgcn_wave_barrier();

    // ---- GEMM3: out = U @ A_flat, K=512 -> 16 WMMAs; U built in registers ----
    v8f acc3 = {};
#pragma unroll
    for (int kt = 0; kt < 16; ++kt) {
      // w[row M, n=2kt] and n=2kt+1 sit in one dword of wLDS
      union { unsigned u; _Float16 h2[2]; } wp;
      wp.u = *(const unsigned*)&wLDS[wave][laneM * N_SYS + kt * 2];
      v16h uf;
#pragma unroll
      for (int i = 0; i < 8; ++i) {
        uf[i]     = wp.h2[0] * dh[i];   // n = 2kt,   d = laneHi*8 + i
        uf[8 + i] = wp.h2[1] * dh[i];   // n = 2kt+1, d = laneHi*8 + i
      }
      v16h bf = *(const v16h*)(wtab + OFF_B3 + kt * 512 + lane * 16);
      acc3 = __builtin_amdgcn_wmma_f32_16x16x32_f16(false, uf, false, bf,
                                                    (short)0, acc3, false, false);
    }

    // ---- store out tile (C layout: M = r + laneHi*8, k = lane&15), coalesced ----
#pragma unroll
    for (int r2 = 0; r2 < 8; ++r2)
      out[(size_t)(b0 + r2 + laneHi * 8) * D_DIM + laneM] = acc3[r2];
  }
}

extern "C" void kernel_launch(void* const* d_in, const int* in_sizes, int n_in,
                              void* d_out, int out_size, void* d_ws, size_t ws_size,
                              hipStream_t stream) {
  (void)n_in; (void)out_size; (void)ws_size;
  const float* x  = (const float*)d_in[0];
  const float* W1 = (const float*)d_in[1];
  const float* b1 = (const float*)d_in[2];
  const float* W2 = (const float*)d_in[3];
  const float* b2 = (const float*)d_in[4];
  const float* Bm = (const float*)d_in[5];
  const float* Cm = (const float*)d_in[6];
  const float* xt = (const float*)d_in[7];
  _Float16* ws = (_Float16*)d_ws;
  float* out = (float*)d_out;

  int rows = in_sizes[0] / D_DIM;   // B = 524288
  int numTiles = rows / 16;         // 32768 row-tiles

  ANDP_prep_frags<<<(WS_HALFS + 255) / 256, 256, 0, stream>>>(W1, W2, Bm, Cm, ws);

  // 1024 blocks x 8 waves = 8192 waves -> 4 tiles/wave; weight table TDM'd to LDS per block
  ANDP_main<<<1024, 256, 0, stream>>>(x, b1, b2, xt, ws, out, numTiles);
}
